// MHA_32066225832318
// MI455X (gfx1250) — matrix operand order repaired and checked by reading, not compiled, not run on hardware
//
#include <hip/hip_runtime.h>
#include <stdint.h>

// MHA forward for B=2, S=2048, D=1024, H=16, d_k=64 on gfx1250 (wave32, WMMA).
// d_out = [ out (2*2048*1024 f32) | attn (2*16*2048*2048 f32) ]
//
// All GEMMs are NT (B source is row-major [N,K]); V is pre-transposed per head
// into f16 so the PV GEMM takes the NT path with direct 32B fragment loads.
// Softmax is register-resident: 1 read + 1 write of the 537MB attn slab.

typedef __attribute__((ext_vector_type(16))) _Float16 v16h;
typedef __attribute__((ext_vector_type(8)))  float    v8f;

// ---- fragment loaders, contiguous k order k = 16*half + s, shared by A and B (wave32) ----

// A: 16x32 (MxK). row = &A[(m0 + lane&15) * lda].
// lanes 0-15 (half=0): V0..3 K=0..7,   V4..7 K=8..15
// lanes16-31 (half=1): V0..3 K=16..23, V4..7 K=24..31
__device__ __forceinline__ v16h load_a_frag(const float* __restrict__ row,
                                            int k, int half) {
  const float* p0 = row + k + 16 * half;
  const float* p1 = p0 + 8;
  v16h a;
#pragma unroll
  for (int r = 0; r < 4; ++r) {
    const float2 f0 = *(const float2*)(p0 + 2 * r);
    const float2 f1 = *(const float2*)(p1 + 2 * r);
    a[2 * r]     = (_Float16)f0.x;
    a[2 * r + 1] = (_Float16)f0.y;
    a[2 * r + 8] = (_Float16)f1.x;
    a[2 * r + 9] = (_Float16)f1.y;
  }
  return a;
}

// B: 32x16 (KxN), NT: B(k,n) = src[n*ldb + k]. row = &B[(n0 + lane&15) * ldb].
// lanes 0-15: K=0..15; lanes 16-31: K=16..31 => 16 contiguous elems at k+16*half.
__device__ __forceinline__ v16h load_b_frag(const float* __restrict__ row,
                                            int k, int half) {
  const float* p = row + k + 16 * half;
  v16h b;
#pragma unroll
  for (int r = 0; r < 8; ++r) {
    const float2 f = *(const float2*)(p + 2 * r);
    b[2 * r]     = (_Float16)f.x;
    b[2 * r + 1] = (_Float16)f.y;
  }
  return b;
}

// Same layout, source already f16: one contiguous 32-byte vector load.
__device__ __forceinline__ v16h load_b_frag(const _Float16* __restrict__ row,
                                            int k, int half) {
  return *(const v16h*)(row + k + 16 * half);
}

// ---- generic batched NT WMMA GEMM ----
// C[m,n] = scale * sum_k A(m,k) * B(n,k) + bias[n]
// Batch over blockIdx.z = b*16 + h with independent b/h strides.
// Block = 128 threads (4 waves stacked in M). Block tile 128x64.
// Wave tile 32x64 = 2x4 fragments -> 8 WMMA per 6 fragment loads.
template <typename BT>
__global__ __launch_bounds__(128)
void gemm_nt(const float* __restrict__ A, int64_t lda, int64_t sAb, int64_t sAh,
             const BT* __restrict__ Bm, int64_t ldb, int64_t sBb, int64_t sBh,
             float* __restrict__ C, int64_t ldc, int64_t sCb, int64_t sCh,
             const float* __restrict__ bias, float scale, int Kdim)
{
  const int z  = blockIdx.z;
  const int bb = z >> 4;
  const int hh = z & 15;
  A  += (int64_t)bb * sAb + (int64_t)hh * sAh;
  Bm += (int64_t)bb * sBb + (int64_t)hh * sBh;
  C  += (int64_t)bb * sCb + (int64_t)hh * sCh;

  const int lane = threadIdx.x & 31;
  const int w    = threadIdx.x >> 5;   // 0..3, waves stacked along M
  const int half = lane >> 4;
  const int l15  = lane & 15;
  const int64_t m0 = (int64_t)blockIdx.y * 128 + w * 32;
  const int64_t n0 = (int64_t)blockIdx.x * 64;

  // Per-lane row pointers, hoisted out of the K loop.
  const float* arow0 = A + (m0 + l15) * lda;
  const float* arow1 = arow0 + 16 * lda;
  const BT* brow[4];
#pragma unroll
  for (int j = 0; j < 4; ++j) brow[j] = Bm + (n0 + 16 * j + l15) * ldb;

  v8f acc[2][4] = {};

  for (int k = 0; k < Kdim; k += 32) {
    // Speculative prefetch 4 K-iterations ahead (silently dropped past the end).
    __builtin_prefetch(arow0 + k + 128, 0, 3);
    __builtin_prefetch(arow1 + k + 128, 0, 3);
#pragma unroll
    for (int j = 0; j < 4; ++j) __builtin_prefetch(brow[j] + k + 128, 0, 3);

    const v16h a0 = load_a_frag(arow0, k, half);
    const v16h a1 = load_a_frag(arow1, k, half);
    v16h b[4];
#pragma unroll
    for (int j = 0; j < 4; ++j) b[j] = load_b_frag(brow[j], k, half);

#pragma unroll
    for (int j = 0; j < 4; ++j) {
      acc[0][j] = __builtin_amdgcn_wmma_f32_16x16x32_f16(false, a0, false, b[j], (short)0, acc[0][j], false, false);
      acc[1][j] = __builtin_amdgcn_wmma_f32_16x16x32_f16(false, a1, false, b[j], (short)0, acc[1][j], false, false);
    }
  }

  // C/D layout: VGPR r -> M = r + 8*half, N = lane&15.
#pragma unroll
  for (int fi = 0; fi < 2; ++fi) {
#pragma unroll
    for (int fj = 0; fj < 4; ++fj) {
      const int64_t nG = n0 + fj * 16 + l15;
      const float bv = bias ? bias[nG] : 0.0f;
#pragma unroll
      for (int r = 0; r < 8; ++r) {
        const int64_t mG = m0 + fi * 16 + r + 8 * half;
        C[mG * ldc + nG] = acc[fi][fj][r] * scale + bv;
      }
    }
  }
}

// ---- per-head V transpose: vt[bh][d][k] (f16) = v[bh][k][d] (f32) ----
__global__ __launch_bounds__(256)
void transpose_v_f16(const float* __restrict__ qkv, _Float16* __restrict__ vt)
{
  const int bh = blockIdx.y;           // b*16 + h
  const int bb = bh >> 4;
  const int hh = bh & 15;
  const int64_t kbase = (int64_t)blockIdx.x * 64;

  const float* V = qkv + (int64_t)bb * 2048 * 3072 + 2048 + hh * 64; // V[k*3072+d]
  _Float16* O = vt + (int64_t)bh * 64 * 2048;                        // O[d][k]

  __shared__ _Float16 tile[64][72];    // [d][k_local], padded rows

  const int t    = threadIdx.x;
  const int krow = t >> 2;             // 0..63
  const int d0   = (t & 3) * 16;       // 0,16,32,48

  const float* src = V + (kbase + krow) * 3072 + d0;
#pragma unroll
  for (int i = 0; i < 16; i += 2) {
    const float2 f = *(const float2*)(src + i);
    tile[d0 + i][krow]     = (_Float16)f.x;
    tile[d0 + i + 1][krow] = (_Float16)f.y;
  }
  __syncthreads();

  const int drow = t >> 2;             // 0..63
  const int k0   = (t & 3) * 16;       // 0,16,32,48
  _Float16* dst = O + (int64_t)drow * 2048 + kbase + k0;
#pragma unroll
  for (int i = 0; i < 16; ++i) dst[i] = tile[drow][k0 + i];
}

// ---- register-resident row softmax (row = 2048, block = 256 threads) ----
// Each thread keeps its 8 elements in registers: 1 read + 1 write of the row.
__global__ __launch_bounds__(256)
void softmax_rows2048(float* __restrict__ data)
{
  const int64_t row = blockIdx.x;
  float* p = data + row * 2048;
  const int t = threadIdx.x;
  float4* q = (float4*)(p + t * 8);

  float4 a = q[0];
  float4 b = q[1];

  __shared__ float red[256];

  float mx = fmaxf(fmaxf(fmaxf(a.x, a.y), fmaxf(a.z, a.w)),
                   fmaxf(fmaxf(b.x, b.y), fmaxf(b.z, b.w)));
  red[t] = mx;
  __syncthreads();
  for (int s = 128; s > 0; s >>= 1) {
    if (t < s) red[t] = fmaxf(red[t], red[t + s]);
    __syncthreads();
  }
  mx = red[0];
  __syncthreads();

  a.x = __expf(a.x - mx); a.y = __expf(a.y - mx);
  a.z = __expf(a.z - mx); a.w = __expf(a.w - mx);
  b.x = __expf(b.x - mx); b.y = __expf(b.y - mx);
  b.z = __expf(b.z - mx); b.w = __expf(b.w - mx);

  red[t] = (a.x + a.y + a.z + a.w) + (b.x + b.y + b.z + b.w);
  __syncthreads();
  for (int s = 128; s > 0; s >>= 1) {
    if (t < s) red[t] += red[t + s];
    __syncthreads();
  }
  const float inv = 1.0f / red[0];

  a.x *= inv; a.y *= inv; a.z *= inv; a.w *= inv;
  b.x *= inv; b.y *= inv; b.z *= inv; b.w *= inv;
  q[0] = a;
  q[1] = b;
}

extern "C" void kernel_launch(void* const* d_in, const int* in_sizes, int n_in,
                              void* d_out, int out_size, void* d_ws, size_t ws_size,
                              hipStream_t stream)
{
  (void)in_sizes; (void)n_in; (void)out_size; (void)ws_size;

  constexpr int64_t B = 2, S = 2048, D = 1024, H = 16, DK = 64;
  constexpr int64_t M   = B * S;          // 4096
  constexpr int64_t OUT = B * S * D;      // 4,194,304

  const float* x    = (const float*)d_in[0];   // [B,S,D]
  const float* Wqkv = (const float*)d_in[1];   // [3D, D]
  const float* bqkv = (const float*)d_in[2];   // [3D]
  const float* Wo   = (const float*)d_in[3];   // [D, D]
  const float* bo   = (const float*)d_in[4];   // [D]

  float* out  = (float*)d_out;                 // [B,S,D]
  float* attn = out + OUT;                     // [B,H,S,S]

  float*    qkv   = (float*)d_ws;              // [B*S, 3D]       50.3 MB
  float*    ohead = qkv + M * 3 * D;           // [B*S, D]        16.8 MB
  _Float16* vt    = (_Float16*)(ohead + OUT);  // [B*H][DK][S]     8.4 MB

  const dim3 blk(128);

  // 1) qkv = x @ Wqkv^T + bqkv            M=4096, N=3072, K=1024
  gemm_nt<float><<<dim3(48, 32, 1), blk, 0, stream>>>(
      x, D, 0, 0,
      Wqkv, D, 0, 0,
      qkv, 3 * D, 0, 0,
      bqkv, 1.0f, (int)D);

  // 2) Vt (f16) = transpose of V per head
  transpose_v_f16<<<dim3(32, 32), dim3(256), 0, stream>>>(qkv, vt);

  // 3) scores = Q @ K^T * 1/sqrt(dk)      per (b,h): M=N=2048, K=64
  gemm_nt<float><<<dim3(32, 16, 32), blk, 0, stream>>>(
      qkv, 3 * D, S * 3 * D, DK,
      qkv + D, 3 * D, S * 3 * D, DK,
      attn, S, H * S * S, S * S,
      nullptr, 0.125f, (int)DK);

  // 4) softmax over last dim, in place (register-resident rows)
  softmax_rows2048<<<dim3((unsigned)(B * H * S)), dim3(256), 0, stream>>>(attn);

  // 5) ohead = attn @ Vt^T (head-merged)  per (b,h): M=2048, N=64, K=2048
  gemm_nt<_Float16><<<dim3(1, 16, 32), blk, 0, stream>>>(
      attn, S, H * S * S, S * S,
      vt, S, H * DK * S, DK * S,
      ohead, D, S * D, DK,
      nullptr, 1.0f, (int)S);

  // 6) out = ohead @ Wo^T + bo            M=4096, N=1024, K=1024
  gemm_nt<float><<<dim3(16, 32, 1), blk, 0, stream>>>(
      ohead, D, 0, 0,
      Wo, D, 0, 0,
      out, D, 0, 0,
      bo, 1.0f, (int)D);
}
